// FusionAttention_5437428596999
// MI455X (gfx1250) — matrix operand order repaired and checked by reading, not compiled, not run on hardware
//
#include <hip/hip_runtime.h>
#include <hip/hip_bf16.h>

// ---------------- constants ----------------
#define FRAMES   8
#define NH       12
#define HD       64
#define CD       768       // channels
#define NO       197       // tokens per frame-batch row
#define BQ       8         // batch after reshape (64/FRAMES)
#define NSEQ     (FRAMES*NO)   // 1576
#define MROWS    (BQ*NSEQ)     // 12608
#define MGATH    204           // gathered K/V tokens
#define MPAD     224           // padded to 7*32

typedef __attribute__((ext_vector_type(16))) _Float16 v16h;
typedef __attribute__((ext_vector_type(8)))  _Float16 v8h;
typedef __attribute__((ext_vector_type(4)))  _Float16 v4h;
typedef __attribute__((ext_vector_type(8)))  float    v8f;
typedef __attribute__((ext_vector_type(4)))  float    v4f;

union V16 { v16h v; v8h h[2]; };

__device__ __forceinline__ v8f wmma16(v16h a, v16h b, v8f c) {
  // D = A(16x32 f16) * B(32x16 f16) + C(16x16 f32)
  return __builtin_amdgcn_wmma_f32_16x16x32_f16(false, a, false, b, (short)0, c,
                                                false, false);
}
__device__ __forceinline__ v8f zero8f() {
  v8f z;
#pragma unroll
  for (int i = 0; i < 8; ++i) z[i] = 0.f;
  return z;
}

// CDNA5 async copy global->LDS (ASYNCcnt path), 16B per lane
__device__ __forceinline__ void async_ld_b128(const void* lds, const void* gptr) {
  unsigned int  ldsoff = (unsigned int)(unsigned long long)lds;  // low 32b = LDS offset
  unsigned long long ga = (unsigned long long)gptr;
  asm volatile("global_load_async_to_lds_b128 %0, %1, off"
               :: "v"(ldsoff), "v"(ga) : "memory");
}
__device__ __forceinline__ void wait_async() {
  asm volatile("s_wait_asynccnt 0" ::: "memory");
}

// ---------------- small prep kernels ----------------
__global__ void cast_f32_to_f16(const float* __restrict__ s,
                                _Float16* __restrict__ d, int n4) {
  int i = blockIdx.x * blockDim.x + threadIdx.x;
  int stride = gridDim.x * blockDim.x;
  for (; i < n4; i += stride) {
    v4f x = ((const v4f*)s)[i];
    v4h y;
#pragma unroll
    for (int k = 0; k < 4; ++k) y[k] = (_Float16)x[k];
    ((v4h*)d)[i] = y;
  }
}

__global__ void zero_f16_kernel(_Float16* __restrict__ d, int n8) {
  int i = blockIdx.x * blockDim.x + threadIdx.x;
  int stride = gridDim.x * blockDim.x;
  v8h z;
#pragma unroll
  for (int k = 0; k < 8; ++k) z[k] = (_Float16)0.f;
  for (; i < n8; i += stride) ((v8h*)d)[i] = z;
}

// idx[j]: source token index for gathered column j; colf[j]: frame of column (-1 = cls)
__global__ void build_idx(int* __restrict__ idxA, int* __restrict__ colf) {
  int j = threadIdx.x;
  if (j >= MGATH) return;
  if (j < FRAMES) { idxA[j] = j * NO; colf[j] = -1; return; }
  int st = FRAMES, i = 0;
#pragma unroll
  for (i = 0; i < FRAMES; ++i) {
    int len = (i < 4) ? 25 : 24;
    if (j < st + len) break;
    st += len;
  }
  int p = j - st;
  idxA[j] = i * NO + 1 + i + 8 * p;
  colf[j] = i;
}

// ---------------- GEMM tiles ----------------
#define BM 128
#define BN 128
#define BK 32

// ---------------- GEMM: Q = Xh * Wq^T  (M=12608, N=768, K=768) ----------------
__global__ __launch_bounds__(256) void gemm_q(const _Float16* __restrict__ A,
                                              const _Float16* __restrict__ W,
                                              _Float16* __restrict__ Q) {
  __shared__ __align__(16) _Float16 As[2][BM * BK];
  __shared__ __align__(16) _Float16 Bs[2][BN * BK];
  const int tid = threadIdx.x, lane = tid & 31, wid = tid >> 5;
  const int l15 = lane & 15, lh = lane >> 4;
  const int waveM = wid >> 1, waveN = wid & 1;
  const int tileM = blockIdx.y * BM, tileN = blockIdx.x * BN;

  v8f acc[2][4];
#pragma unroll
  for (int i = 0; i < 2; ++i)
#pragma unroll
    for (int j = 0; j < 4; ++j) acc[i][j] = zero8f();

  auto stage = [&](int buf, int kk) {
#pragma unroll
    for (int c = tid; c < (BM * BK / 8); c += 256) {
      int row = c >> 2, kc = (c & 3) * 8;
      int gr = tileM + row; if (gr >= MROWS) gr = MROWS - 1;
      async_ld_b128(&As[buf][row * BK + kc], &A[(size_t)gr * CD + kk + kc]);
      async_ld_b128(&Bs[buf][row * BK + kc], &W[(size_t)(tileN + row) * CD + kk + kc]);
    }
  };

  stage(0, 0);
  wait_async();
  __syncthreads();

  for (int k0 = 0; k0 < CD; k0 += BK) {
    const int buf = (k0 / BK) & 1;
    if (k0 + BK < CD) stage(buf ^ 1, k0 + BK);   // prefetch next tile (async)

    V16 a[2], b[4];
#pragma unroll
    for (int i = 0; i < 2; ++i) {
      int mr = waveM * 32 + i * 16 + l15;
      a[i].h[0] = *(const v8h*)&As[buf][mr * BK + lh * 8];
      a[i].h[1] = *(const v8h*)&As[buf][mr * BK + 16 + lh * 8];
    }
#pragma unroll
    for (int j = 0; j < 4; ++j) {
      int nc = waveN * 64 + j * 16 + l15;
      b[j].h[0] = *(const v8h*)&Bs[buf][nc * BK + lh * 8];
      b[j].h[1] = *(const v8h*)&Bs[buf][nc * BK + 16 + lh * 8];
    }
#pragma unroll
    for (int i = 0; i < 2; ++i)
#pragma unroll
      for (int j = 0; j < 4; ++j) acc[i][j] = wmma16(a[i].v, b[j].v, acc[i][j]);

    wait_async();
    __syncthreads();
  }
#pragma unroll
  for (int i = 0; i < 2; ++i)
#pragma unroll
    for (int j = 0; j < 4; ++j) {
      int col = tileN + waveN * 64 + j * 16 + l15;
#pragma unroll
      for (int r = 0; r < 8; ++r) {
        int row = tileM + waveM * 32 + i * 16 + lh * 8 + r;
        if (row < MROWS) Q[(size_t)row * CD + col] = (_Float16)acc[i][j][r];
      }
    }
}

// ------- GEMM: gathered K/V = Xh[idx] * Wkv^T  (per batch, M=204, N=1536) -------
__global__ __launch_bounds__(256) void gemm_kv(const _Float16* __restrict__ A,
                                               const _Float16* __restrict__ Wkv,
                                               const int* __restrict__ idxA,
                                               _Float16* __restrict__ Kb,
                                               _Float16* __restrict__ Vb) {
  __shared__ __align__(16) _Float16 As[2][BM * BK];
  __shared__ __align__(16) _Float16 Bs[2][BN * BK];
  const int tid = threadIdx.x, lane = tid & 31, wid = tid >> 5;
  const int l15 = lane & 15, lh = lane >> 4;
  const int waveM = wid >> 1, waveN = wid & 1;
  const int tileM = blockIdx.y * BM, tileN = blockIdx.x * BN;
  const int b = blockIdx.z;

  v8f acc[2][4];
#pragma unroll
  for (int i = 0; i < 2; ++i)
#pragma unroll
    for (int j = 0; j < 4; ++j) acc[i][j] = zero8f();

  auto stage = [&](int buf, int kk) {
#pragma unroll
    for (int c = tid; c < (BM * BK / 8); c += 256) {
      int row = c >> 2, kc = (c & 3) * 8;
      int j = tileM + row; if (j >= MGATH) j = MGATH - 1;
      int gr = b * NSEQ + idxA[j];
      async_ld_b128(&As[buf][row * BK + kc], &A[(size_t)gr * CD + kk + kc]);
      async_ld_b128(&Bs[buf][row * BK + kc], &Wkv[(size_t)(tileN + row) * CD + kk + kc]);
    }
  };

  stage(0, 0);
  wait_async();
  __syncthreads();

  for (int k0 = 0; k0 < CD; k0 += BK) {
    const int buf = (k0 / BK) & 1;
    if (k0 + BK < CD) stage(buf ^ 1, k0 + BK);

    V16 a[2], bf[4];
#pragma unroll
    for (int i = 0; i < 2; ++i) {
      int mr = waveM * 32 + i * 16 + l15;
      a[i].h[0] = *(const v8h*)&As[buf][mr * BK + lh * 8];
      a[i].h[1] = *(const v8h*)&As[buf][mr * BK + 16 + lh * 8];
    }
#pragma unroll
    for (int j = 0; j < 4; ++j) {
      int nc = waveN * 64 + j * 16 + l15;
      bf[j].h[0] = *(const v8h*)&Bs[buf][nc * BK + lh * 8];
      bf[j].h[1] = *(const v8h*)&Bs[buf][nc * BK + 16 + lh * 8];
    }
#pragma unroll
    for (int i = 0; i < 2; ++i)
#pragma unroll
      for (int j = 0; j < 4; ++j) acc[i][j] = wmma16(a[i].v, bf[j].v, acc[i][j]);

    wait_async();
    __syncthreads();
  }
  // scatter-store into (b,h,j,d) layouts for K and V
#pragma unroll
  for (int i = 0; i < 2; ++i)
#pragma unroll
    for (int j = 0; j < 4; ++j) {
      int col = tileN + waveN * 64 + j * 16 + l15;   // 0..1535
#pragma unroll
      for (int r = 0; r < 8; ++r) {
        int jj = tileM + waveM * 32 + i * 16 + lh * 8 + r;
        if (jj < MGATH) {
          _Float16 val = (_Float16)acc[i][j][r];
          if (col < CD) {
            int h = col >> 6, d = col & 63;
            Kb[(((size_t)b * NH + h) * MPAD + jj) * HD + d] = val;
          } else {
            int c2 = col - CD;
            int h = c2 >> 6, d = c2 & 63;
            Vb[(((size_t)b * NH + h) * MPAD + jj) * HD + d] = val;
          }
        }
      }
    }
}

// ---------------- flash attention over gathered K/V ----------------
__global__ __launch_bounds__(256) void attn_kernel(const _Float16* __restrict__ Qb,
                                                   const _Float16* __restrict__ Kb,
                                                   const _Float16* __restrict__ Vb,
                                                   const int* __restrict__ colf,
                                                   _Float16* __restrict__ Ob) {
  __shared__ __align__(16) _Float16 Kc[32 * HD];      // [m][d]
  __shared__ __align__(16) _Float16 Vt[HD * 32];      // [d][m] (transposed at staging)
  __shared__ __align__(16) _Float16 Ps[8 * 16 * 32];  // per-wave P scratch
  const int tid = threadIdx.x, lane = tid & 31, wid = tid >> 5;
  const int l15 = lane & 15, lh = lane >> 4;
  const int bh = blockIdx.y, b = bh / NH, h = bh % NH;
  const int qbase = blockIdx.x * 128 + wid * 16;
  const _Float16* Kp = Kb + (size_t)bh * MPAD * HD;
  const _Float16* Vp = Vb + (size_t)bh * MPAD * HD;

  // Q fragments (16 rows x 64 d  -> two 16x32 A-frags)
  V16 qa[2];
  {
    int row = qbase + l15; if (row >= NSEQ) row = NSEQ - 1;
    const _Float16* qp = Qb + (size_t)(b * NSEQ + row) * CD + h * HD;
#pragma unroll
    for (int kk = 0; kk < 2; ++kk) {
      qa[kk].h[0] = *(const v8h*)&qp[kk * 32 + lh * 8];
      qa[kk].h[1] = *(const v8h*)&qp[kk * 32 + 16 + lh * 8];
    }
  }
  float mrow[8], lrow[8];
  int fq[8];
  v8f o[4];
#pragma unroll
  for (int dt = 0; dt < 4; ++dt) o[dt] = zero8f();
#pragma unroll
  for (int r = 0; r < 8; ++r) {
    mrow[r] = -1e30f; lrow[r] = 0.f;
    int qr = qbase + lh * 8 + r; if (qr >= NSEQ) qr = NSEQ - 1;
    fq[r] = qr / NO;
  }
  const float sc = 0.125f;  // 64^-0.5

  for (int ch = 0; ch < MPAD / 32; ++ch) {
    __syncthreads();
    {   // cooperative stage: K chunk via async copy, V chunk transposed via regs
      int m = tid >> 3, d0 = (tid & 7) * 8;
      async_ld_b128(&Kc[m * HD + d0], &Kp[(size_t)(ch * 32 + m) * HD + d0]);
      v8h vv = *(const v8h*)&Vp[(size_t)(ch * 32 + m) * HD + d0];
#pragma unroll
      for (int i = 0; i < 8; ++i) Vt[(d0 + i) * 32 + m] = vv[i];
      wait_async();
    }
    __syncthreads();

    // S = Q K^T  (16 x 32)
    v8f s[2]; s[0] = zero8f(); s[1] = zero8f();
#pragma unroll
    for (int nt = 0; nt < 2; ++nt) {
      int mcol = nt * 16 + l15;
      V16 kb0, kb1;
      kb0.h[0] = *(const v8h*)&Kc[mcol * HD + lh * 8];
      kb0.h[1] = *(const v8h*)&Kc[mcol * HD + 16 + lh * 8];
      kb1.h[0] = *(const v8h*)&Kc[mcol * HD + 32 + lh * 8];
      kb1.h[1] = *(const v8h*)&Kc[mcol * HD + 32 + 16 + lh * 8];
      s[nt] = wmma16(qa[0].v, kb0.v, s[nt]);
      s[nt] = wmma16(qa[1].v, kb1.v, s[nt]);
    }

    // scale + multiplicative mask (1.0 / 0.8) + OOB -> -inf
    float chmax[8];
#pragma unroll
    for (int r = 0; r < 8; ++r) chmax[r] = -1e30f;
#pragma unroll
    for (int nt = 0; nt < 2; ++nt) {
      int j = ch * 32 + nt * 16 + l15;
      int cf = colf[j < MGATH ? j : 0];
      bool oob = (j >= MGATH);
#pragma unroll
      for (int r = 0; r < 8; ++r) {
        float v = s[nt][r] * sc;
        v *= (cf < 0 || cf == fq[r]) ? 1.0f : 0.8f;
        if (oob) v = -1e30f;
        s[nt][r] = v;
        chmax[r] = fmaxf(chmax[r], v);
      }
    }
#pragma unroll
    for (int off = 8; off >= 1; off >>= 1)
#pragma unroll
      for (int r = 0; r < 8; ++r)
        chmax[r] = fmaxf(chmax[r], __shfl_xor(chmax[r], off, 16));

    float alpha[8], chsum[8];
#pragma unroll
    for (int r = 0; r < 8; ++r) {
      float nm = fmaxf(mrow[r], chmax[r]);
      alpha[r] = __expf(mrow[r] - nm);
      mrow[r] = nm;
      chsum[r] = 0.f;
    }
    _Float16* Pw = &Ps[wid * 16 * 32];
#pragma unroll
    for (int nt = 0; nt < 2; ++nt)
#pragma unroll
      for (int r = 0; r < 8; ++r) {
        float p = __expf(s[nt][r] - mrow[r]);
        chsum[r] += p;
        Pw[(lh * 8 + r) * 32 + nt * 16 + l15] = (_Float16)p;
      }
#pragma unroll
    for (int off = 8; off >= 1; off >>= 1)
#pragma unroll
      for (int r = 0; r < 8; ++r) chsum[r] += __shfl_xor(chsum[r], off, 16);
#pragma unroll
    for (int r = 0; r < 8; ++r) lrow[r] = lrow[r] * alpha[r] + chsum[r];
#pragma unroll
    for (int dt = 0; dt < 4; ++dt)
#pragma unroll
      for (int r = 0; r < 8; ++r) o[dt][r] *= alpha[r];

    // read P back in A-fragment layout, O += P * V
    V16 pa;
    pa.h[0] = *(const v8h*)&Pw[l15 * 32 + lh * 8];
    pa.h[1] = *(const v8h*)&Pw[l15 * 32 + 16 + lh * 8];
#pragma unroll
    for (int dt = 0; dt < 4; ++dt) {
      int dcol = dt * 16 + l15;
      V16 vb;
      vb.h[0] = *(const v8h*)&Vt[dcol * 32 + lh * 8];
      vb.h[1] = *(const v8h*)&Vt[dcol * 32 + 16 + lh * 8];
      o[dt] = wmma16(pa.v, vb.v, o[dt]);
    }
  }
  // normalize + store (b, n, h*64+d) as f16 for the projection GEMM
#pragma unroll
  for (int r = 0; r < 8; ++r) {
    int row = qbase + lh * 8 + r;
    if (row >= NSEQ) continue;
    float inv = 1.0f / lrow[r];
    _Float16* op = Ob + (size_t)(b * NSEQ + row) * CD + h * HD;
#pragma unroll
    for (int dt = 0; dt < 4; ++dt) op[dt * 16 + l15] = (_Float16)(o[dt][r] * inv);
  }
}

// ---------------- projection: out = Ob * Wp^T + bias (f32 out) ----------------
__global__ __launch_bounds__(256) void gemm_proj(const _Float16* __restrict__ A,
                                                 const _Float16* __restrict__ W,
                                                 const float* __restrict__ bias,
                                                 float* __restrict__ out) {
  __shared__ __align__(16) _Float16 As[2][BM * BK];
  __shared__ __align__(16) _Float16 Bs[2][BN * BK];
  const int tid = threadIdx.x, lane = tid & 31, wid = tid >> 5;
  const int l15 = lane & 15, lh = lane >> 4;
  const int waveM = wid >> 1, waveN = wid & 1;
  const int tileM = blockIdx.y * BM, tileN = blockIdx.x * BN;

  v8f acc[2][4];
#pragma unroll
  for (int i = 0; i < 2; ++i)
#pragma unroll
    for (int j = 0; j < 4; ++j) acc[i][j] = zero8f();

  auto stage = [&](int buf, int kk) {
#pragma unroll
    for (int c = tid; c < (BM * BK / 8); c += 256) {
      int row = c >> 2, kc = (c & 3) * 8;
      int gr = tileM + row; if (gr >= MROWS) gr = MROWS - 1;
      async_ld_b128(&As[buf][row * BK + kc], &A[(size_t)gr * CD + kk + kc]);
      async_ld_b128(&Bs[buf][row * BK + kc], &W[(size_t)(tileN + row) * CD + kk + kc]);
    }
  };

  stage(0, 0);
  wait_async();
  __syncthreads();

  for (int k0 = 0; k0 < CD; k0 += BK) {
    const int buf = (k0 / BK) & 1;
    if (k0 + BK < CD) stage(buf ^ 1, k0 + BK);

    V16 a[2], bf[4];
#pragma unroll
    for (int i = 0; i < 2; ++i) {
      int mr = waveM * 32 + i * 16 + l15;
      a[i].h[0] = *(const v8h*)&As[buf][mr * BK + lh * 8];
      a[i].h[1] = *(const v8h*)&As[buf][mr * BK + 16 + lh * 8];
    }
#pragma unroll
    for (int j = 0; j < 4; ++j) {
      int nc = waveN * 64 + j * 16 + l15;
      bf[j].h[0] = *(const v8h*)&Bs[buf][nc * BK + lh * 8];
      bf[j].h[1] = *(const v8h*)&Bs[buf][nc * BK + 16 + lh * 8];
    }
#pragma unroll
    for (int i = 0; i < 2; ++i)
#pragma unroll
      for (int j = 0; j < 4; ++j) acc[i][j] = wmma16(a[i].v, bf[j].v, acc[i][j]);

    wait_async();
    __syncthreads();
  }
#pragma unroll
  for (int i = 0; i < 2; ++i)
#pragma unroll
    for (int j = 0; j < 4; ++j) {
      int col = tileN + waveN * 64 + j * 16 + l15;
      float bsv = bias[col];
#pragma unroll
      for (int r = 0; r < 8; ++r) {
        int row = tileM + waveM * 32 + i * 16 + lh * 8 + r;
        if (row < MROWS) out[(size_t)row * CD + col] = acc[i][j][r] + bsv;
      }
    }
}

// ---------------- host launcher ----------------
extern "C" void kernel_launch(void* const* d_in, const int* in_sizes, int n_in,
                              void* d_out, int out_size, void* d_ws, size_t ws_size,
                              hipStream_t stream) {
  const float* x      = (const float*)d_in[0];
  const float* qkv_w  = (const float*)d_in[1];
  const float* proj_w = (const float*)d_in[2];
  const float* proj_b = (const float*)d_in[3];
  float* out = (float*)d_out;

  char* ws = (char*)d_ws;
  size_t off = 0;
  auto carve = [&](size_t bytes) {
    void* p = ws + off;
    off += (bytes + 255) & ~(size_t)255;
    return p;
  };
  _Float16* xh   = (_Float16*)carve((size_t)MROWS * CD * 2);
  _Float16* wqkv = (_Float16*)carve((size_t)3 * CD * CD * 2);
  _Float16* wp   = (_Float16*)carve((size_t)CD * CD * 2);
  _Float16* Qb   = (_Float16*)carve((size_t)MROWS * CD * 2);
  _Float16* Kb   = (_Float16*)carve((size_t)BQ * NH * MPAD * HD * 2);
  _Float16* Vb   = (_Float16*)carve((size_t)BQ * NH * MPAD * HD * 2);
  _Float16* Ob   = (_Float16*)carve((size_t)MROWS * CD * 2);
  int* idxA = (int*)carve(256 * 4);
  int* colf = (int*)carve(256 * 4);

  build_idx<<<1, 256, 0, stream>>>(idxA, colf);
  cast_f32_to_f16<<<2048, 256, 0, stream>>>(x, xh, MROWS * CD / 4);
  cast_f32_to_f16<<<1024, 256, 0, stream>>>(qkv_w, wqkv, 3 * CD * CD / 4);
  cast_f32_to_f16<<<512, 256, 0, stream>>>(proj_w, wp, CD * CD / 4);
  zero_f16_kernel<<<512, 256, 0, stream>>>(Kb, BQ * NH * MPAD * HD / 8);
  zero_f16_kernel<<<512, 256, 0, stream>>>(Vb, BQ * NH * MPAD * HD / 8);

  // Q = Xh * Wq^T
  gemm_q<<<dim3(CD / BN, (MROWS + BM - 1) / BM), 256, 0, stream>>>(xh, wqkv, Qb);
  // gathered K/V = Xh[idx] * Wkv^T
  gemm_kv<<<dim3(2 * CD / BN, (MGATH + BM - 1) / BM, BQ), 256, 0, stream>>>(
      xh, wqkv + (size_t)CD * CD, idxA, Kb, Vb);
  // flash attention
  attn_kernel<<<dim3((NSEQ + 127) / 128, BQ * NH), 256, 0, stream>>>(Qb, Kb, Vb,
                                                                     colf, Ob);
  // projection + bias
  gemm_proj<<<dim3(CD / BN, (MROWS + BM - 1) / BM), 256, 0, stream>>>(Ob, wp,
                                                                      proj_b, out);
}
